// GATNet_45157286150393
// MI455X (gfx1250) — compile-verified, matrix operand order checked by reading, not run on hardware
//
#include <hip/hip_runtime.h>
#include <hip/hip_bf16.h>

typedef __attribute__((ext_vector_type(16))) _Float16 v16h;
typedef __attribute__((ext_vector_type(8)))  _Float16 v8h;
typedef __attribute__((ext_vector_type(8)))  float    v8f;

#define NN 512
#define HH 64
#define NEDGE 8192
#define GTHRESH 0.3f

// ---------------- WMMA helpers (gfx1250, wave32) ----------------
// 16-bit A-matrix 16x32 fragment from row-major [16, ldk] at column k0.
// lane<16: M=lane, K = k0+{0..7} (v0..3) and k0+{16..23} (v4..7)
// lane>=16: M=lane-16, K = k0+8+{0..7} and k0+24+{0..7}
__device__ static inline v16h ldfrag(const _Float16* base, int ldk, int k0, int lane) {
  const _Float16* p = base + (size_t)(lane & 15) * ldk + k0 + ((lane >> 4) << 3);
  v8h lo = *(const v8h*)(p);
  v8h hi = *(const v8h*)(p + 16);
  v16h f;
#pragma unroll
  for (int t = 0; t < 8; ++t) { f[t] = lo[t]; f[t + 8] = hi[t]; }
  return f;
}
// C layout: vgpr v, lane l -> row = v + ((l>>4)<<3), col = l&15

// ---------------- Stage A: per-node prep ----------------
__global__ void k_node_prep(const int* hids, const float* emb_h, const float* src_emb,
                            const float* dst_emb, const float* pg1emb, const float* g1,
                            const float* b1, const float* pg2emb, const float* g2,
                            const float* b2, const float* pgw, const float* epw,
                            const float* epb, const float* ep3b, const float* p2w,
                            const float* p2b, const float* Wh1, const float* as1,
                            const float* ad1, float* hfeat, _Float16* hp2_16,
                            _Float16* u16, _Float16* p2_16, float* sloc, float* dloc,
                            float* z1, float* asrc1, float* adst1) {
  int node = blockIdx.x, k = threadIdx.x, hid = hids[node];
  __shared__ float se[64], ss[64], sd[64], sz[64];
  float hf = emb_h[hid * 64 + k];
  hfeat[node * 64 + k] = hf; se[k] = hf;
  ss[k] = src_emb[hid * 64 + k];
  sd[k] = dst_emb[hid * 64 + k];
  float t2 = pg2emb[hid * 64 + k] * g2[k] + b2[k];
  hp2_16[node * 64 + k] = (_Float16)fmaxf(t2, 0.f);
  float ua = 0.f;
  for (int c = 0; c < 64; ++c) {
    float x = pg1emb[(size_t)hid * 4096 + c * 64 + k];
    x = x * g1[c * 64 + k] + b1[c * 64 + k];
    ua += pgw[c] * fmaxf(x, 0.f);
  }
  u16[node * 64 + k] = (_Float16)ua;
  __syncthreads();
  float a = 0.f, bs = 0.f, cs = 0.f, ds = 0.f;
  for (int t = 0; t < 64; ++t) {
    a  += se[t] * p2w[t * 64 + k];
    bs += ss[t] * epw[t * 64 + k];
    cs += sd[t] * epw[(64 + t) * 64 + k];
    ds += se[t] * Wh1[t * 64 + k];
  }
  p2_16[node * 64 + k] = (_Float16)(a + p2b[k]);
  sloc[node * 64 + k] = bs + epb[k] + ep3b[k];  // fold both edge biases here
  dloc[node * 64 + k] = cs;
  sz[k] = ds; z1[node * 64 + k] = ds;
  __syncthreads();
  if (k < 8) {
    float sa = 0.f, sdd = 0.f;
    for (int dd = 0; dd < 8; ++dd) {
      sa  += sz[k * 8 + dd] * as1[k * 8 + dd];
      sdd += sz[k * 8 + dd] * ad1[k * 8 + dd];
    }
    asrc1[node * 8 + k] = sa; adst1[node * 8 + k] = sdd;
  }
}

__global__ void k_p1(const float* hfeat, const float* p1w, const float* p1b, float* P1) {
  int node = blockIdx.x, tid = threadIdx.x;
  __shared__ float hf[64];
  if (tid < 64) hf[tid] = hfeat[node * 64 + tid];
  __syncthreads();
  for (int r = 0; r < 16; ++r) {
    int m = tid + 256 * r;
    float acc = p1b[m];
    for (int t = 0; t < 64; ++t) acc += hf[t] * p1w[(size_t)t * 4096 + m];
    P1[(size_t)node * 4096 + m] = acc;
  }
}

__global__ void k_q(const float* P1, const float* W3, _Float16* Q16) {
  int node = blockIdx.x, tid = threadIdx.x;
  __shared__ float p1s[4096];
  __shared__ float w3s[4096];
  for (int r = 0; r < 16; ++r) {
    p1s[tid + 256 * r] = P1[(size_t)node * 4096 + tid + 256 * r];
    w3s[tid + 256 * r] = W3[tid + 256 * r];
  }
  __syncthreads();
  for (int r = 0; r < 16; ++r) {
    int o = tid + 256 * r, d = o >> 6, k = o & 63;
    float acc = 0.f;
    for (int c = 0; c < 64; ++c) acc += w3s[c * 64 + d] * p1s[c * 64 + k];
    Q16[(size_t)node * 4096 + o] = (_Float16)acc;
  }
}

// ---------------- S = u @ hp2^T (+bias, zero diag) ----------------
__global__ void k_sgemm(const _Float16* U, const _Float16* HP2, const float* pgb, float* S) {
  int lane = threadIdx.x & 31, wave = threadIdx.x >> 5;
  int t = blockIdx.x * 8 + wave;  // 0..1023 tiles of 32x32 grid
  int mt = t >> 5, nt = t & 31;
  v16h a0 = ldfrag(U + mt * 16 * 64, 64, 0, lane);
  v16h a1 = ldfrag(U + mt * 16 * 64, 64, 32, lane);
  v16h b0 = ldfrag(HP2 + nt * 16 * 64, 64, 0, lane);
  v16h b1 = ldfrag(HP2 + nt * 16 * 64, 64, 32, lane);
  v8f c = {};
  c = __builtin_amdgcn_wmma_f32_16x16x32_f16(false, a0, false, b0, (short)0, c, false, false);
  c = __builtin_amdgcn_wmma_f32_16x16x32_f16(false, a1, false, b1, (short)0, c, false, false);
  float bias = pgb[0];
  int mh = (lane >> 4) << 3, col = nt * 16 + (lane & 15);
#pragma unroll
  for (int q = 0; q < 8; ++q) {
    int row = mt * 16 + q + mh;
    float val = c[q] + bias;
    if (row == col) val = 0.f;
    S[row * NN + col] = val;
  }
}

// ---------------- softmax stats + adjacency mask ----------------
__global__ void k_row_stats(const float* S, float* rmax, float* rsum) {
  int i = blockIdx.x, tid = threadIdx.x;
  __shared__ float red[256];
  float v0 = S[i * NN + tid], v1 = S[i * NN + tid + 256];
  red[tid] = fmaxf(v0, v1); __syncthreads();
  for (int s = 128; s > 0; s >>= 1) { if (tid < s) red[tid] = fmaxf(red[tid], red[tid + s]); __syncthreads(); }
  float m = red[0]; __syncthreads();
  red[tid] = expf(v0 - m) + expf(v1 - m); __syncthreads();
  for (int s = 128; s > 0; s >>= 1) { if (tid < s) red[tid] += red[tid + s]; __syncthreads(); }
  if (tid == 0) { rmax[i] = m; rsum[i] = red[0]; }
}
__global__ void k_col_stats(const float* S, float* cmax, float* csum) {
  int j = blockIdx.x, tid = threadIdx.x;
  __shared__ float red[256];
  float v0 = S[tid * NN + j], v1 = S[(tid + 256) * NN + j];
  red[tid] = fmaxf(v0, v1); __syncthreads();
  for (int s = 128; s > 0; s >>= 1) { if (tid < s) red[tid] = fmaxf(red[tid], red[tid + s]); __syncthreads(); }
  float m = red[0]; __syncthreads();
  red[tid] = expf(v0 - m) + expf(v1 - m); __syncthreads();
  for (int s = 128; s > 0; s >>= 1) { if (tid < s) red[tid] += red[tid + s]; __syncthreads(); }
  if (tid == 0) { cmax[j] = m; csum[j] = red[0]; }
}
__global__ void k_zero_mask(unsigned int* m) { m[blockIdx.x * 256 + threadIdx.x] = 0u; }
__global__ void k_adj(const int* ei, unsigned char* mask) {
  int k = blockIdx.x * 256 + threadIdx.x;
  mask[ei[k] * NN + ei[NEDGE + k]] = 1;
}
__global__ void k_thresh(const float* S, const float* rmax, const float* rsum,
                         const float* cmax, const float* csum, unsigned char* mask) {
  int i = blockIdx.x, tid = threadIdx.x;
  float rm = rmax[i], rs = rsum[i];
  for (int r = 0; r < 2; ++r) {
    int j = tid + 256 * r;
    float v = S[i * NN + j];
    float p = (expf(v - cmax[j]) / csum[j]) * (expf(v - rm) / rs);
    if (p > GTHRESH) mask[i * NN + j] = 1;
  }
}

// ---------------- E-GEMM: e[(i,d),j] = Q @ P2^T, fused epilogue ----------------
// Per-lane C elements are 8 consecutive d at fixed (i,j): pack one b128 store.
__global__ void k_egemm(const _Float16* Q, const _Float16* P2, const float* sloc,
                        const float* dloc, const float* beg, const float* beb,
                        _Float16* E16) {
  int lane = threadIdx.x & 31, wave = threadIdx.x >> 5;
  int mt = blockIdx.x;  // 0..2047 (M = 32768)
  const _Float16* abase = Q + (size_t)mt * 16 * 64;
  v16h a0 = ldfrag(abase, 64, 0, lane), a1 = ldfrag(abase, 64, 32, lane);
  int mh = (lane >> 4) << 3, cn = lane & 15;
  int i = (mt * 16) >> 6;            // constant: 16-row tile stays in one node i
  int d0 = ((mt * 16) & 63) + mh;    // 8 consecutive d per lane
#pragma unroll
  for (int q = 0; q < 4; ++q) {
    int nt = wave * 4 + q;
    const _Float16* bbase = P2 + nt * 16 * 64;
    v16h b0 = ldfrag(bbase, 64, 0, lane), b1 = ldfrag(bbase, 64, 32, lane);
    v8f c = {};
    c = __builtin_amdgcn_wmma_f32_16x16x32_f16(false, a0, false, b0, (short)0, c, false, false);
    c = __builtin_amdgcn_wmma_f32_16x16x32_f16(false, a1, false, b1, (short)0, c, false, false);
    int j = nt * 16 + cn;
    v8h pack;
#pragma unroll
    for (int vv = 0; vv < 8; ++vv) {
      int d = d0 + vv;
      float val = c[vv] + sloc[i * 64 + d] + dloc[j * 64 + d];
      val = fmaxf(val * beg[d] + beb[d], 0.f);
      pack[vv] = (_Float16)val;
    }
    *(v8h*)(E16 + ((size_t)i * NN + j) * 64 + d0) = pack;
  }
}

// ---------------- small weight prep ----------------
__global__ void k_wet(const float* We1, _Float16* wet) {
  for (int o = threadIdx.x; o < 4096; o += 256) {
    int n = o >> 6, k = o & 63;
    wet[o] = (_Float16)We1[k * 64 + n];
  }
}
__global__ void k_v(const float* We2, const float* ae2, float* v) {
  int k = threadIdx.x;
  float a = 0.f;
  for (int d = 0; d < 64; ++d) a += We2[k * 64 + d] * ae2[d];
  v[k] = a;
}

// ---------------- ZE-GEMM: ze = e @ We1 ----------------
// Roles swapped vs naive: A = We^T tile (M = hd), B = e rows (N = (i,j)).
// Per-lane C elements are 8 consecutive hd at fixed (i,j): pack one b128 store.
__global__ void k_zegemm(const _Float16* E16, const _Float16* wet, _Float16* ZE16) {
  int lane = threadIdx.x & 31, wave = threadIdx.x >> 5;
  int mt = blockIdx.x * 8 + wave;  // 0..16383 column-tile over r = i*512+j
  const _Float16* bbase = E16 + (size_t)mt * 16 * 64;
  v16h b0 = ldfrag(bbase, 64, 0, lane), b1 = ldfrag(bbase, 64, 32, lane);
  int mh = (lane >> 4) << 3, cn = lane & 15;
  size_t r = (size_t)mt * 16 + cn;
#pragma unroll
  for (int q = 0; q < 4; ++q) {
    const _Float16* abase = wet + q * 16 * 64;
    v16h a0 = ldfrag(abase, 64, 0, lane), a1 = ldfrag(abase, 64, 32, lane);
    v8f c = {};
    c = __builtin_amdgcn_wmma_f32_16x16x32_f16(false, a0, false, b0, (short)0, c, false, false);
    c = __builtin_amdgcn_wmma_f32_16x16x32_f16(false, a1, false, b1, (short)0, c, false, false);
    int hd0 = q * 16 + mh;
    v8h pack;
#pragma unroll
    for (int vv = 0; vv < 8; ++vv) pack[vv] = (_Float16)c[vv];
    *(v8h*)(ZE16 + r * 64 + hd0) = pack;
  }
}

// ---------------- GAT1 attention ----------------
__global__ void k_score1(const _Float16* ZE16, const float* asrc1, const float* adst1,
                         const float* a_e, const unsigned char* mask, float* sT) {
  int j = blockIdx.x, i = blockIdx.y * 256 + threadIdx.x;
  __shared__ float ae[64], ad[8];
  if (threadIdx.x < 64) ae[threadIdx.x] = a_e[threadIdx.x];
  if (threadIdx.x < 8) ad[threadIdx.x] = adst1[j * 8 + threadIdx.x];
  __syncthreads();
  const v8h* zr = (const v8h*)(ZE16 + ((size_t)i * NN + j) * 64);
  bool msk = mask[i * NN + j] != 0;
#pragma unroll
  for (int h = 0; h < 8; ++h) {
    v8h zh = zr[h];
    float acc = asrc1[i * 8 + h] + ad[h];
#pragma unroll
    for (int d = 0; d < 8; ++d) acc += (float)zh[d] * ae[h * 8 + d];
    float lr = acc > 0.f ? acc : 0.2f * acc;
    sT[((size_t)j * 8 + h) * NN + i] = msk ? lr : -1e9f;
  }
}
__global__ void k_softmax1(float* sT, const unsigned char* mask) {
  int col = blockIdx.x, j = col >> 3, tid = threadIdx.x;
  __shared__ float red[256];
  float* p = sT + (size_t)col * NN;
  float v0 = p[tid], v1 = p[tid + 256];
  red[tid] = fmaxf(v0, v1); __syncthreads();
  for (int s = 128; s > 0; s >>= 1) { if (tid < s) red[tid] = fmaxf(red[tid], red[tid + s]); __syncthreads(); }
  float m = red[0]; __syncthreads();
  float e0 = expf(v0 - m), e1 = expf(v1 - m);
  red[tid] = e0 + e1; __syncthreads();
  for (int s = 128; s > 0; s >>= 1) { if (tid < s) red[tid] += red[tid + s]; __syncthreads(); }
  float inv = 1.f / red[0];
  float m0 = mask[tid * NN + j] ? 1.f : 0.f;
  float m1 = mask[(tid + 256) * NN + j] ? 1.f : 0.f;
  p[tid] = e0 * inv * m0; p[tid + 256] = e1 * inv * m1;
}
__global__ void k_agg1(const float* alphaT, const float* z1, const float* hfeat,
                       const float* g, const float* b, float* hx) {
  int j = blockIdx.x, t = threadIdx.x;
  __shared__ float al[4096];
  for (int m = t; m < 4096; m += 64) al[m] = alphaT[(size_t)j * 4096 + m];
  __syncthreads();
  int h = t >> 3;
  float acc = 0.f;
  for (int i = 0; i < NN; ++i) acc += al[h * NN + i] * z1[i * 64 + t];
  acc += hfeat[j * 64 + t];
  acc = acc * g[t] + b[t];
  hx[j * 64 + t] = fmaxf(acc, 0.f);
}

// ---------------- GAT2 (ze2 GEMM algebraically eliminated) ----------------
__global__ void k_gat2prep(const float* hx, const float* Wh2, const float* as2,
                           const float* ad2, float* z2, float* asrc2, float* adst2) {
  int node = blockIdx.x, k = threadIdx.x;
  __shared__ float hr[64], zr[64];
  hr[k] = hx[node * 64 + k]; __syncthreads();
  float acc = 0.f;
  for (int t = 0; t < 64; ++t) acc += hr[t] * Wh2[t * 64 + k];
  zr[k] = acc; z2[node * 64 + k] = acc;
  __syncthreads();
  if (k == 0) { float a = 0.f; for (int d = 0; d < 64; ++d) a += zr[d] * as2[d]; asrc2[node] = a; }
  if (k == 1) { float a = 0.f; for (int d = 0; d < 64; ++d) a += zr[d] * ad2[d]; adst2[node] = a; }
}
__global__ void k_score2(const _Float16* ZE16, const _Float16* E16, const float* g1e,
                         const float* b1e, const float* v, const float* asrc2,
                         const float* adst2, const unsigned char* mask, float* s2T) {
  int j = blockIdx.x, i = blockIdx.y * 256 + threadIdx.x;
  __shared__ float g[64], bb[64], vv[64];
  if (threadIdx.x < 64) {
    g[threadIdx.x] = g1e[threadIdx.x];
    bb[threadIdx.x] = b1e[threadIdx.x];
    vv[threadIdx.x] = v[threadIdx.x];
  }
  __syncthreads();
  const v8h* zr = (const v8h*)(ZE16 + ((size_t)i * NN + j) * 64);
  const v8h* er = (const v8h*)(E16 + ((size_t)i * NN + j) * 64);
  float acc = 0.f;
#pragma unroll
  for (int q = 0; q < 8; ++q) {
    v8h zq = zr[q], eq = er[q];
#pragma unroll
    for (int d = 0; d < 8; ++d) {
      int k = q * 8 + d;
      float ex = ((float)zq[d] + (float)eq[d]) * g[k] + bb[k];
      acc += fmaxf(ex, 0.f) * vv[k];
    }
  }
  acc += asrc2[i] + adst2[j];
  float lr = acc > 0.f ? acc : 0.2f * acc;
  s2T[(size_t)j * NN + i] = mask[i * NN + j] ? lr : -1e9f;
}
__global__ void k_softmax2(float* s2T, const unsigned char* mask) {
  int j = blockIdx.x, tid = threadIdx.x;
  __shared__ float red[256];
  float* p = s2T + (size_t)j * NN;
  float v0 = p[tid], v1 = p[tid + 256];
  red[tid] = fmaxf(v0, v1); __syncthreads();
  for (int s = 128; s > 0; s >>= 1) { if (tid < s) red[tid] = fmaxf(red[tid], red[tid + s]); __syncthreads(); }
  float m = red[0]; __syncthreads();
  float e0 = expf(v0 - m), e1 = expf(v1 - m);
  red[tid] = e0 + e1; __syncthreads();
  for (int s = 128; s > 0; s >>= 1) { if (tid < s) red[tid] += red[tid + s]; __syncthreads(); }
  float inv = 1.f / red[0];
  float m0 = mask[tid * NN + j] ? 1.f : 0.f;
  float m1 = mask[(tid + 256) * NN + j] ? 1.f : 0.f;
  p[tid] = e0 * inv * m0; p[tid + 256] = e1 * inv * m1;
}
__global__ void k_agg2(const float* alpha2T, const float* z2, const float* hx,
                       const float* g, const float* b, float* hx2) {
  int j = blockIdx.x, t = threadIdx.x;
  __shared__ float al[512];
  for (int m = t; m < 512; m += 64) al[m] = alpha2T[(size_t)j * NN + m];
  __syncthreads();
  float acc = 0.f;
  for (int i = 0; i < NN; ++i) acc += al[i] * z2[i * 64 + t];
  acc += hx[j * 64 + t];
  acc = acc * g[t] + b[t];
  hx2[j * 64 + t] = fmaxf(acc, 0.f);
}

// ---------------- MLP head ----------------
__global__ void k_mlp(const float* hx2, const float* w1, const float* b1, const float* w2,
                      const float* b2, const float* w3, const float* b3, float* out) {
  __shared__ float W1[2048], B1[32], W2[512], B2[16], W3s[160], B3[10];
  int tid = threadIdx.x;
  for (int m = tid; m < 2048; m += 256) W1[m] = w1[m];
  for (int m = tid; m < 512; m += 256) W2[m] = w2[m];
  if (tid < 32) B1[tid] = b1[tid];
  if (tid < 16) B2[tid] = b2[tid];
  if (tid < 160) W3s[tid] = w3[tid];
  if (tid < 10) B3[tid] = b3[tid];
  __syncthreads();
  int n = blockIdx.x * 256 + tid;
  float x[64];
  for (int t = 0; t < 64; ++t) x[t] = hx2[n * 64 + t];
  float a1[32];
  for (int c = 0; c < 32; ++c) {
    float a = B1[c];
    for (int t = 0; t < 64; ++t) a += x[t] * W1[t * 32 + c];
    a1[c] = fmaxf(a, 0.f);
  }
  float a2[16];
  for (int c = 0; c < 16; ++c) {
    float a = B2[c];
    for (int t = 0; t < 32; ++t) a += a1[t] * W2[t * 16 + c];
    a2[c] = fmaxf(a, 0.f);
  }
  for (int c = 0; c < 10; ++c) {
    float a = B3[c];
    for (int t = 0; t < 16; ++t) a += a2[t] * W3s[t * 10 + c];
    out[n * 10 + c] = a;
  }
}

// ---------------- workspace layout (bytes) ----------------
static constexpr size_t oHP2 = 0;                     // 512*64 f16
static constexpr size_t oU = oHP2 + 65536;
static constexpr size_t oP2 = oU + 65536;
static constexpr size_t oWET = oP2 + 65536;           // 64*64 f16
static constexpr size_t oV = oWET + 8192;             // 64 f32 (padded)
static constexpr size_t oRMAX = oV + 1024;
static constexpr size_t oRSUM = oRMAX + 2048;
static constexpr size_t oCMAX = oRSUM + 2048;
static constexpr size_t oCSUM = oCMAX + 2048;
static constexpr size_t oAS1 = oCSUM + 2048;          // 512*8 f32
static constexpr size_t oAD1 = oAS1 + 16384;
static constexpr size_t oAS2 = oAD1 + 16384;          // 512 f32
static constexpr size_t oAD2 = oAS2 + 2048;
static constexpr size_t oHFEAT = oAD2 + 2048;         // 512*64 f32 blocks
static constexpr size_t oSLOC = oHFEAT + 131072;
static constexpr size_t oDLOC = oSLOC + 131072;
static constexpr size_t oZ1 = oDLOC + 131072;
static constexpr size_t oHX = oZ1 + 131072;
static constexpr size_t oZ2 = oHX + 131072;
static constexpr size_t oHX2 = oZ2 + 131072;
static constexpr size_t oMASK = oHX2 + 131072;        // 512*512 u8
static constexpr size_t oS = oMASK + 262144;          // 512*512 f32
static constexpr size_t oS2T = oS + 1048576;          // 512*512 f32
static constexpr size_t oST = oS2T + 1048576;         // 512*8*512 f32
static constexpr size_t oP1 = oST + 8388608;          // 512*4096 f32
static constexpr size_t oQ = oP1 + 8388608;           // 512*4096 f16
static constexpr size_t oE16 = oQ + 4194304;          // 512*512*64 f16
static constexpr size_t oZE16 = oE16 + 33554432;      // 512*512*64 f16

extern "C" void kernel_launch(void* const* d_in, const int* in_sizes, int n_in,
                              void* d_out, int out_size, void* d_ws, size_t ws_size,
                              hipStream_t stream) {
  (void)in_sizes; (void)n_in; (void)out_size; (void)ws_size;
  const int* hids = (const int*)d_in[0];
  const int* ei = (const int*)d_in[1];
  const float* emb_h = (const float*)d_in[2];
  const float* src_emb = (const float*)d_in[3];
  const float* dst_emb = (const float*)d_in[4];
  const float* pg1emb = (const float*)d_in[5];
  const float* g1 = (const float*)d_in[6];
  const float* b1 = (const float*)d_in[7];
  const float* pg2emb = (const float*)d_in[8];
  const float* g2 = (const float*)d_in[9];
  const float* b2 = (const float*)d_in[10];
  const float* pgw = (const float*)d_in[11];
  const float* pgb = (const float*)d_in[12];
  const float* epw = (const float*)d_in[13];
  const float* epb = (const float*)d_in[14];
  const float* p1w = (const float*)d_in[15];
  const float* p1b = (const float*)d_in[16];
  const float* p2w = (const float*)d_in[17];
  const float* p2b = (const float*)d_in[18];
  const float* w3 = (const float*)d_in[19];   // edge_proj3_w
  const float* w3b = (const float*)d_in[20];  // edge_proj3_b
  const float* beg = (const float*)d_in[21];  // top bn_e_g
  const float* beb = (const float*)d_in[22];
  const float* Wh1 = (const float*)d_in[23];
  const float* We1 = (const float*)d_in[24];
  const float* as1 = (const float*)d_in[25];
  const float* ad1 = (const float*)d_in[26];
  const float* ae1 = (const float*)d_in[27];
  const float* bh1g = (const float*)d_in[28];
  const float* bh1b = (const float*)d_in[29];
  const float* be1g = (const float*)d_in[30];
  const float* be1b = (const float*)d_in[31];
  const float* Wh2 = (const float*)d_in[32];
  const float* We2 = (const float*)d_in[33];
  const float* as2 = (const float*)d_in[34];
  const float* ad2 = (const float*)d_in[35];
  const float* ae2 = (const float*)d_in[36];
  const float* bh2g = (const float*)d_in[37];
  const float* bh2b = (const float*)d_in[38];
  const float* mw1 = (const float*)d_in[41];
  const float* mb1 = (const float*)d_in[42];
  const float* mw2 = (const float*)d_in[43];
  const float* mb2 = (const float*)d_in[44];
  const float* mw3 = (const float*)d_in[45];
  const float* mb3 = (const float*)d_in[46];
  float* out = (float*)d_out;
  char* ws = (char*)d_ws;

  _Float16* HP2 = (_Float16*)(ws + oHP2);
  _Float16* U = (_Float16*)(ws + oU);
  _Float16* P2 = (_Float16*)(ws + oP2);
  _Float16* WET = (_Float16*)(ws + oWET);
  float* V = (float*)(ws + oV);
  float* RMAX = (float*)(ws + oRMAX); float* RSUM = (float*)(ws + oRSUM);
  float* CMAX = (float*)(ws + oCMAX); float* CSUM = (float*)(ws + oCSUM);
  float* AS1 = (float*)(ws + oAS1); float* AD1 = (float*)(ws + oAD1);
  float* AS2 = (float*)(ws + oAS2); float* AD2 = (float*)(ws + oAD2);
  float* HFEAT = (float*)(ws + oHFEAT);
  float* SLOC = (float*)(ws + oSLOC); float* DLOC = (float*)(ws + oDLOC);
  float* Z1 = (float*)(ws + oZ1); float* HX = (float*)(ws + oHX);
  float* Z2 = (float*)(ws + oZ2); float* HX2 = (float*)(ws + oHX2);
  unsigned char* MASK = (unsigned char*)(ws + oMASK);
  float* S = (float*)(ws + oS); float* S2T = (float*)(ws + oS2T);
  float* ST = (float*)(ws + oST);
  float* P1 = (float*)(ws + oP1);
  _Float16* Q = (_Float16*)(ws + oQ);
  _Float16* E16 = (_Float16*)(ws + oE16);
  _Float16* ZE16 = (_Float16*)(ws + oZE16);

  k_zero_mask<<<256, 256, 0, stream>>>((unsigned int*)MASK);
  k_node_prep<<<512, 64, 0, stream>>>(hids, emb_h, src_emb, dst_emb, pg1emb, g1, b1,
                                      pg2emb, g2, b2, pgw, epw, epb, w3b, p2w, p2b,
                                      Wh1, as1, ad1, HFEAT, HP2, U, P2, SLOC, DLOC,
                                      Z1, AS1, AD1);
  k_p1<<<512, 256, 0, stream>>>(HFEAT, p1w, p1b, P1);
  k_q<<<512, 256, 0, stream>>>(P1, w3, Q);
  k_sgemm<<<128, 256, 0, stream>>>(U, HP2, pgb, S);
  k_row_stats<<<512, 256, 0, stream>>>(S, RMAX, RSUM);
  k_col_stats<<<512, 256, 0, stream>>>(S, CMAX, CSUM);
  k_adj<<<32, 256, 0, stream>>>(ei, MASK);
  k_thresh<<<512, 256, 0, stream>>>(S, RMAX, RSUM, CMAX, CSUM, MASK);
  k_wet<<<1, 256, 0, stream>>>(We1, WET);
  k_v<<<1, 64, 0, stream>>>(We2, ae2, V);
  k_egemm<<<2048, 256, 0, stream>>>(Q, P2, SLOC, DLOC, beg, beb, E16);
  k_zegemm<<<2048, 256, 0, stream>>>(E16, WET, ZE16);
  k_score1<<<dim3(512, 2), 256, 0, stream>>>(ZE16, AS1, AD1, ae1, MASK, ST);
  k_softmax1<<<4096, 256, 0, stream>>>(ST, MASK);
  k_agg1<<<512, 64, 0, stream>>>(ST, Z1, HFEAT, bh1g, bh1b, HX);
  k_gat2prep<<<512, 64, 0, stream>>>(HX, Wh2, as2, ad2, Z2, AS2, AD2);
  k_score2<<<dim3(512, 2), 256, 0, stream>>>(ZE16, E16, be1g, be1b, V, AS2, AD2, MASK, S2T);
  k_softmax2<<<512, 256, 0, stream>>>(S2T, MASK);
  k_agg2<<<512, 64, 0, stream>>>(S2T, Z2, HX, bh2g, bh2b, HX2);
  k_mlp<<<2, 256, 0, stream>>>(HX2, mw1, mb1, mw2, mb2, mw3, mb3, out);
}
